// DNCCell_46334107189521
// MI455X (gfx1250) — matrix operand order repaired and checked by reading, not compiled, not run on hardware
//
#include <hip/hip_runtime.h>
#include <hip/hip_bf16.h>
#include <stdint.h>

typedef __attribute__((ext_vector_type(16))) _Float16 v16h;
typedef __attribute__((ext_vector_type(8)))  _Float16 v8h;
typedef __attribute__((ext_vector_type(4)))  _Float16 v4h;
typedef __attribute__((ext_vector_type(8)))  float    v8f;

#define B_      256
#define H_      1024
#define M_      128
#define A_      2048
#define P_      926
#define P_PAD   960
#define P_READ_ 134
#define PW_OFF  536

__device__ __forceinline__ float sigmoidf_(float x) { return 1.0f / (1.0f + expf(-x)); }
__device__ __forceinline__ float softplusf_(float x) { return (x > 20.0f) ? x : log1pf(expf(x)); }

// ---------------- CDNA5 async global->LDS helpers --------------------------
__device__ __forceinline__ void g2l_async_b128(const _Float16* g, _Float16* l) {
    unsigned int laddr = (unsigned int)(unsigned long long)(uintptr_t)l;
    asm volatile("global_load_async_to_lds_b128 %0, %1, off" :: "v"(laddr), "v"(g) : "memory");
}
__device__ __forceinline__ void wait_async0() {
    asm volatile("s_wait_asynccnt 0" ::: "memory");
}

// ---------------- packing / conversion kernels -----------------------------
__global__ void pack_act_kernel(const float* __restrict__ x, const float* __restrict__ rvp,
                                const float* __restrict__ hp, _Float16* __restrict__ act) {
    int idx = blockIdx.x * blockDim.x + threadIdx.x;   // 256*512 threads, 4 elems each
    int b = idx >> 9, kq = idx & 511;
    int k = kq * 4;
    const float* src;
    if (k < 512)       src = x   + (size_t)b * 512  + k;
    else if (k < 1024) src = rvp + (size_t)b * 512  + (k - 512);
    else               src = hp  + (size_t)b * 1024 + (k - 1024);
    float4 v = *(const float4*)src;
    v4h o = { (_Float16)v.x, (_Float16)v.y, (_Float16)v.z, (_Float16)v.w };
    *(v4h*)(act + (size_t)b * 2048 + k) = o;
}

// transpose+convert: src f32 [K][N]  ->  dst f16 rows [Nrows][Kdst] at column offset koff
__global__ void transpose_conv_kernel(const float* __restrict__ src, _Float16* __restrict__ dst,
                                      int K, int N, int Kdst, int koff, int Nrows) {
    __shared__ float tile[32][33];
    const int tx = threadIdx.x, ty = threadIdx.y;
    const int k0 = blockIdx.y * 32, n0 = blockIdx.x * 32;
    for (int j = ty; j < 32; j += 8) {
        int k = k0 + j, n = n0 + tx;
        tile[j][tx] = (k < K && n < N) ? src[(size_t)k * N + n] : 0.0f;
    }
    __syncthreads();
    for (int j = ty; j < 32; j += 8) {
        int n = n0 + j, k = k0 + tx;
        if (n < Nrows && k < K)
            dst[(size_t)n * Kdst + koff + k] = (_Float16)tile[tx][j];
    }
}

__global__ void pad_bias_kernel(const float* __restrict__ src, float* __restrict__ dst, int n, int npad) {
    int i = blockIdx.x * blockDim.x + threadIdx.x;
    if (i < npad) dst[i] = (i < n) ? src[i] : 0.0f;
}

__global__ void lstm_gates_kernel(const float* __restrict__ z, const float* __restrict__ c_prev,
                                  _Float16* __restrict__ h16) {
    int idx = blockIdx.x * blockDim.x + threadIdx.x;   // 256*256 threads, 4 elems each
    int b = idx >> 8, jj = idx & 255;
    const float4* zr = (const float4*)(z + (size_t)b * 4096);
    float4 ig = zr[jj], fg = zr[256 + jj], gg = zr[512 + jj], og = zr[768 + jj];
    float4 cp = ((const float4*)c_prev)[(size_t)(b << 8) + jj];
    float c0 = sigmoidf_(fg.x) * cp.x + sigmoidf_(ig.x) * tanhf(gg.x);
    float c1 = sigmoidf_(fg.y) * cp.y + sigmoidf_(ig.y) * tanhf(gg.y);
    float c2 = sigmoidf_(fg.z) * cp.z + sigmoidf_(ig.z) * tanhf(gg.z);
    float c3 = sigmoidf_(fg.w) * cp.w + sigmoidf_(ig.w) * tanhf(gg.w);
    v4h o = { (_Float16)(sigmoidf_(og.x) * tanhf(c0)), (_Float16)(sigmoidf_(og.y) * tanhf(c1)),
              (_Float16)(sigmoidf_(og.z) * tanhf(c2)), (_Float16)(sigmoidf_(og.w) * tanhf(c3)) };
    *(v4h*)(h16 + (size_t)b * 1024 + jj * 4) = o;
}

// ---------------- WMMA GEMM ------------------------------------------------
// C[Md][Nd] = A[Md][Kd](f16) @ Bt[Nd][Kd]^T(f16) + bias  (+ addm if HASADD)
#define BM 128
#define BN 64
#define BK 32
#define LSTR 40   // halfs per LDS row: 80B, 16B-aligned

union U8f { v8f v; float e[8]; };

template <bool HASADD>
__global__ __launch_bounds__(256) void gemm_wmma_kernel(
    const _Float16* __restrict__ A, const _Float16* __restrict__ Bt,
    const float* __restrict__ bias, const float* __restrict__ addm,
    float* __restrict__ C, int Kd, int Nd)
{
    __shared__ __align__(16) _Float16 As[2][BM][LSTR];
    __shared__ __align__(16) _Float16 Bs[2][BN][LSTR];

    const int t = threadIdx.x;
    const int wave = t >> 5, lane = t & 31;
    const int wr = wave & 3, wc = wave >> 2;           // 4x2 wave grid, 32x32 per wave
    const int rowbase = blockIdx.y * BM;
    const int colbase = blockIdx.x * BN;

    // staging: A = 128 rows x 2 segs(16 halfs); B = 64 rows x 4 segs(8 halfs)
    const int am = t >> 1, aseg = (t & 1) * 16;
    const int bn = t >> 2, bseg = (t & 3) * 8;

    auto issue = [&](int buf, int k0) {
        const _Float16* ga = A + (size_t)(rowbase + am) * Kd + k0 + aseg;
        g2l_async_b128(ga,     &As[buf][am][aseg]);
        g2l_async_b128(ga + 8, &As[buf][am][aseg + 8]);
        g2l_async_b128(Bt + (size_t)(colbase + bn) * Kd + k0 + bseg, &Bs[buf][bn][bseg]);
    };

    v8f acc00 = {0,0,0,0,0,0,0,0}, acc01 = acc00, acc10 = acc00, acc11 = acc00;

    issue(0, 0);
    wait_async0();
    __syncthreads();

    const int nk = Kd / BK;
    const int half = lane >> 4;
    const int l15 = lane & 15;
    const int ar0 = wr * 32 + l15, ar1 = ar0 + 16;
    const int bc0 = wc * 32 + l15, bc1 = bc0 + 16;

    for (int ks = 0; ks < nk; ++ks) {
        const int cur = ks & 1;
        if (ks + 1 < nk) issue(cur ^ 1, (ks + 1) * BK);

        // A fragment (ISA 16-bit A 16x32): two contiguous 16B runs per lane
        const _Float16* a0p = &As[cur][ar0][0];
        const _Float16* a1p = &As[cur][ar1][0];
        v8h a0lo = *(const v8h*)(a0p + 8 * half);
        v8h a0hi = *(const v8h*)(a0p + 16 + 8 * half);
        v8h a1lo = *(const v8h*)(a1p + 8 * half);
        v8h a1hi = *(const v8h*)(a1p + 16 + 8 * half);
        v16h a0 = __builtin_shufflevector(a0lo, a0hi, 0,1,2,3,4,5,6,7,8,9,10,11,12,13,14,15);
        v16h a1 = __builtin_shufflevector(a1lo, a1hi, 0,1,2,3,4,5,6,7,8,9,10,11,12,13,14,15);
        // B fragment: lane = column, same K order as A: two contiguous 16B runs per lane
        const _Float16* b0p = &Bs[cur][bc0][8 * half];
        const _Float16* b1p = &Bs[cur][bc1][8 * half];
        v8h b0lo = *(const v8h*)(b0p), b0hi = *(const v8h*)(b0p + 16);
        v8h b1lo = *(const v8h*)(b1p), b1hi = *(const v8h*)(b1p + 16);
        v16h b0 = __builtin_shufflevector(b0lo, b0hi, 0,1,2,3,4,5,6,7,8,9,10,11,12,13,14,15);
        v16h b1 = __builtin_shufflevector(b1lo, b1hi, 0,1,2,3,4,5,6,7,8,9,10,11,12,13,14,15);

        acc00 = __builtin_amdgcn_wmma_f32_16x16x32_f16(false, a0, false, b0, (short)0, acc00, false, false);
        acc01 = __builtin_amdgcn_wmma_f32_16x16x32_f16(false, a0, false, b1, (short)0, acc01, false, false);
        acc10 = __builtin_amdgcn_wmma_f32_16x16x32_f16(false, a1, false, b0, (short)0, acc10, false, false);
        acc11 = __builtin_amdgcn_wmma_f32_16x16x32_f16(false, a1, false, b1, (short)0, acc11, false, false);

        wait_async0();
        __syncthreads();
    }

    // Epilogue (no guards: all dims padded). C/D layout: VGPR v -> M = v + 8*half.
    const int gc0 = colbase + bc0;
    const int gc1 = colbase + bc1;
    const float bias0 = bias[gc0];
    const float bias1 = bias[gc1];
    U8f u00, u01, u10, u11;
    u00.v = acc00; u01.v = acc01; u10.v = acc10; u11.v = acc11;
#pragma unroll
    for (int v = 0; v < 8; ++v) {
        int grow0 = rowbase + wr * 32 + v + 8 * half;
        int grow1 = grow0 + 16;
        float r00 = u00.e[v] + bias0, r01 = u01.e[v] + bias1;
        float r10 = u10.e[v] + bias0, r11 = u11.e[v] + bias1;
        if (HASADD) {
            r00 += addm[(size_t)grow0 * Nd + gc0];
            r01 += addm[(size_t)grow0 * Nd + gc1];
            r10 += addm[(size_t)grow1 * Nd + gc0];
            r11 += addm[(size_t)grow1 * Nd + gc1];
        }
        C[(size_t)grow0 * Nd + gc0] = r00;
        C[(size_t)grow0 * Nd + gc1] = r01;
        C[(size_t)grow1 * Nd + gc0] = r10;
        C[(size_t)grow1 * Nd + gc1] = r11;
    }
}

// ---------------- block reduction (1024 threads, wave32) -------------------
__device__ __forceinline__ float block_reduce(float v, float* red, bool domax) {
#pragma unroll
    for (int o = 16; o > 0; o >>= 1) {
        float u = __shfl_xor(v, o, 32);
        v = domax ? fmaxf(v, u) : (v + u);
    }
    const int wid = threadIdx.x >> 5, ln = threadIdx.x & 31;
    __syncthreads();
    if (ln == 0) red[wid] = v;
    __syncthreads();
    if (wid == 0) {
        float x2 = red[ln];
#pragma unroll
        for (int o = 16; o > 0; o >>= 1) {
            float u = __shfl_xor(x2, o, 32);
            x2 = domax ? fmaxf(x2, u) : (x2 + u);
        }
        if (ln == 0) red[0] = x2;
    }
    __syncthreads();
    return red[0];
}

// ---------------- content addressing: one streaming pass over memory -------
__global__ __launch_bounds__(1024) void addressing_kernel(
    const float* __restrict__ p, const float* __restrict__ rwp,
    const float* __restrict__ wwp, const float* __restrict__ mem,
    float* __restrict__ w_read, float* __restrict__ w_write)
{
    __shared__ float kS[5 * 128];
    __shared__ float hp[5 * 6];      // beta, g, gamma, s0, s1, s2
    __shared__ float knorm[5];
    __shared__ float wg[2048];
    __shared__ float red[32];

    const int b = blockIdx.x;
    const int t = threadIdx.x;
    const float* pb = p + (size_t)b * P_PAD;

    if (t < 640) {
        int h = t >> 7, m = t & 127;
        kS[t] = pb[(h < 4) ? h * P_READ_ + m : PW_OFF + m];
    } else if (t < 645) {
        int h = t - 640;
        int base = (h < 4) ? h * P_READ_ : PW_OFF;
        hp[h * 6 + 0] = softplusf_(pb[base + 128]);
        hp[h * 6 + 1] = sigmoidf_(pb[base + 129]);
        hp[h * 6 + 2] = 1.0f + softplusf_(pb[base + 130]);
        float s0 = pb[base + 131], s1 = pb[base + 132], s2 = pb[base + 133];
        float mx = fmaxf(s0, fmaxf(s1, s2));
        float e0 = expf(s0 - mx), e1 = expf(s1 - mx), e2 = expf(s2 - mx);
        float inv = 1.0f / (e0 + e1 + e2);
        hp[h * 6 + 3] = e0 * inv; hp[h * 6 + 4] = e1 * inv; hp[h * 6 + 5] = e2 * inv;
    }
    __syncthreads();
    if (t < 5) {
        float s = 0.0f;
        for (int m = 0; m < 128; ++m) { float v = kS[t * 128 + m]; s += v * v; }
        knorm[t] = sqrtf(s);
    }
    __syncthreads();

    // thread owns cols 2t, 2t+1 (float2 loads); fused 5-head dots + col sumsq
    float d0[5] = {0, 0, 0, 0, 0}, d1[5] = {0, 0, 0, 0, 0};
    float ss0 = 0.0f, ss1 = 0.0f;
    const float* mb = mem + (size_t)b * M_ * A_;
    for (int m = 0; m < 128; ++m) {
        float2 v = ((const float2*)(mb + (size_t)m * A_))[t];
        ss0 += v.x * v.x; ss1 += v.y * v.y;
#pragma unroll
        for (int h = 0; h < 5; ++h) {
            float kv = kS[h * 128 + m];
            d0[h] += kv * v.x; d1[h] += kv * v.y;
        }
    }
    float mn0 = sqrtf(ss0), mn1 = sqrtf(ss1);
    const int a0 = 2 * t, a1 = 2 * t + 1;

    for (int h = 0; h < 5; ++h) {
        float beta = hp[h * 6 + 0], g = hp[h * 6 + 1], gam = hp[h * 6 + 2];
        float s0 = hp[h * 6 + 3], s1 = hp[h * 6 + 4], s2 = hp[h * 6 + 5];
        float kn = knorm[h];
        float x0 = beta * (d0[h] / (kn * mn0 + 1e-8f));
        float x1 = beta * (d1[h] / (kn * mn1 + 1e-8f));
        float mx = block_reduce(fmaxf(x0, x1), red, true);
        float e0 = expf(x0 - mx), e1 = expf(x1 - mx);
        float sum = block_reduce(e0 + e1, red, false);
        float inv = 1.0f / sum;
        const float* r = (h < 4) ? (rwp + ((size_t)b * 4 + h) * A_) : (wwp + (size_t)b * A_);
        float2 wp = ((const float2*)r)[t];
        float wg0 = g * (e0 * inv) + (1.0f - g) * wp.x;
        float wg1 = g * (e1 * inv) + (1.0f - g) * wp.y;
        wg[a0] = wg0; wg[a1] = wg1;
        __syncthreads();
        // w_s[a] = s0*wg[a+1] + s1*wg[a] + s2*wg[a-1]  (mod 2048)
        float ws0 = s0 * wg1 + s1 * wg0 + s2 * wg[(a0 + 2047) & 2047];
        float ws1 = s0 * wg[(a1 + 1) & 2047] + s1 * wg1 + s2 * wg0;
        float pw0 = powf(ws0 + 1e-12f, gam);
        float pw1 = powf(ws1 + 1e-12f, gam);
        float psum = block_reduce(pw0 + pw1, red, false);
        float pinv = 1.0f / (psum + 1e-8f);
        float* o = (h < 4) ? (w_read + ((size_t)b * 4 + h) * A_) : (w_write + (size_t)b * A_);
        float2 ov = { pw0 * pinv, pw1 * pinv };
        ((float2*)o)[t] = ov;
        __syncthreads();
    }
}

// ---------------- fused memory update + read_vec ---------------------------
__global__ __launch_bounds__(1024) void mem_update_kernel(
    const float* __restrict__ mem, const float* __restrict__ p,
    const float* __restrict__ w_read, const float* __restrict__ w_write,
    float* __restrict__ mem_out, _Float16* __restrict__ rv16)
{
    __shared__ __align__(16) float wrS[4 * 2048];
    __shared__ __align__(16) float wwS[2048];
    __shared__ float erS[128];
    __shared__ float adS[128];

    const int b = blockIdx.x, t = threadIdx.x;
    {
        const float4* wrg = (const float4*)(w_read + (size_t)b * 4 * 2048);
        float4* wrs4 = (float4*)wrS;
        for (int i = t; i < 2048; i += 1024) wrs4[i] = wrg[i];
        if (t < 512) ((float4*)wwS)[t] = ((const float4*)(w_write + (size_t)b * 2048))[t];
        if (t < 128) {
            const float* pw = p + (size_t)b * P_PAD + PW_OFF;
            erS[t] = sigmoidf_(pw[134 + t]);
            adS[t] = pw[262 + t];
        }
    }
    __syncthreads();

    const int wave = t >> 5, lane = t & 31;     // wave owns rows m = wave*4..wave*4+3
    float acc[4][4];
#pragma unroll
    for (int mi = 0; mi < 4; ++mi)
#pragma unroll
        for (int r = 0; r < 4; ++r) acc[mi][r] = 0.0f;

    const size_t mbase = (size_t)b * M_ * A_;
    const float4* wwS4 = (const float4*)wwS;
#pragma unroll
    for (int mi = 0; mi < 4; ++mi) {
        const int m = wave * 4 + mi;
        const size_t rb = mbase + (size_t)m * A_;
        const float4* mrow = (const float4*)(mem + rb);
        float4* orow = (float4*)(mem_out + rb);
        const float em = erS[m], am = adS[m];
        for (int ch = 0; ch < 16; ++ch) {
            int idx = (ch << 5) + lane;          // float4 index within row (512 per row)
            float4 v = mrow[idx];
            float4 ww = wwS4[idx];
            float4 nv;
            nv.x = v.x * (1.0f - em * ww.x) + am * ww.x;
            nv.y = v.y * (1.0f - em * ww.y) + am * ww.y;
            nv.z = v.z * (1.0f - em * ww.z) + am * ww.z;
            nv.w = v.w * (1.0f - em * ww.w) + am * ww.w;
            orow[idx] = nv;
#pragma unroll
            for (int r = 0; r < 4; ++r) {
                float4 wr4 = ((const float4*)(wrS + r * 2048))[idx];
                acc[mi][r] += v.x * wr4.x + v.y * wr4.y + v.z * wr4.z + v.w * wr4.w;
            }
        }
    }
#pragma unroll
    for (int mi = 0; mi < 4; ++mi)
#pragma unroll
        for (int r = 0; r < 4; ++r) {
            float s = acc[mi][r];
#pragma unroll
            for (int o = 16; o > 0; o >>= 1) s += __shfl_xor(s, o, 32);
            if (lane == 0) rv16[(size_t)b * 512 + r * 128 + wave * 4 + mi] = (_Float16)s;
        }
}

// ---------------- launcher -------------------------------------------------
extern "C" void kernel_launch(void* const* d_in, const int* in_sizes, int n_in,
                              void* d_out, int out_size, void* d_ws, size_t ws_size,
                              hipStream_t stream) {
    const float* x            = (const float*)d_in[0];
    const float* h_prev       = (const float*)d_in[1];
    const float* c_prev       = (const float*)d_in[2];
    const float* read_w_prev  = (const float*)d_in[3];
    const float* write_w_prev = (const float*)d_in[4];
    const float* memory_prev  = (const float*)d_in[5];
    const float* read_vec_prev= (const float*)d_in[6];
    const float* Wx           = (const float*)d_in[7];
    const float* Wh           = (const float*)d_in[8];
    const float* b_lstm       = (const float*)d_in[9];
    const float* Wo_h         = (const float*)d_in[10];
    const float* bo_h         = (const float*)d_in[11];
    const float* Wint         = (const float*)d_in[12];
    const float* b_int        = (const float*)d_in[13];
    const float* Wout         = (const float*)d_in[14];
    const float* b_out        = (const float*)d_in[15];

    float* out_final = (float*)d_out;
    float* out_mem   = (float*)d_out + (size_t)B_ * 512;

    char* w = (char*)d_ws;
    size_t off = 0;
    auto carve = [&](size_t bytes) -> void* {
        void* pp = w + off;
        off = (off + bytes + 255) & ~(size_t)255;
        return pp;
    };
    _Float16* act16  = (_Float16*)carve((size_t)256 * 2048 * 2);
    _Float16* wlT    = (_Float16*)carve((size_t)4096 * 2048 * 2);  // [N=4096][K=2048]
    _Float16* wintT  = (_Float16*)carve((size_t)960 * 1024 * 2);   // [N=960][K=1024]
    _Float16* wohT   = (_Float16*)carve((size_t)512 * 1024 * 2);   // [N=512][K=1024]
    _Float16* woutT  = (_Float16*)carve((size_t)512 * 512 * 2);    // [N=512][K=512]
    float*    bint960= (float*)   carve((size_t)960 * 4);
    float*    z      = (float*)   carve((size_t)256 * 4096 * 4);
    _Float16* h16    = (_Float16*)carve((size_t)256 * 1024 * 2);
    float*    outh   = (float*)   carve((size_t)256 * 512 * 4);
    float*    pbuf   = (float*)   carve((size_t)256 * 960 * 4);
    float*    wread  = (float*)   carve((size_t)256 * 4 * 2048 * 4);
    float*    wwrite = (float*)   carve((size_t)256 * 2048 * 4);
    _Float16* rv16   = (_Float16*)carve((size_t)256 * 512 * 2);

    dim3 t32x8(32, 8);
    // 0) pack activations + transpose/convert weights to f16 [N][K], pad bias
    pack_act_kernel<<<512, 256, 0, stream>>>(x, read_vec_prev, h_prev, act16);
    transpose_conv_kernel<<<dim3(128, 32), t32x8, 0, stream>>>(Wx,   wlT,   1024, 4096, 2048, 0,    4096);
    transpose_conv_kernel<<<dim3(128, 32), t32x8, 0, stream>>>(Wh,   wlT,   1024, 4096, 2048, 1024, 4096);
    transpose_conv_kernel<<<dim3(30, 32),  t32x8, 0, stream>>>(Wint, wintT, 1024, 926,  1024, 0,    960);
    transpose_conv_kernel<<<dim3(16, 32),  t32x8, 0, stream>>>(Wo_h, wohT,  1024, 512,  1024, 0,    512);
    transpose_conv_kernel<<<dim3(16, 16),  t32x8, 0, stream>>>(Wout, woutT, 512,  512,  512,  0,    512);
    pad_bias_kernel<<<4, 256, 0, stream>>>(b_int, bint960, 926, 960);

    // 1) LSTM GEMM -> z
    gemm_wmma_kernel<false><<<dim3(4096 / BN, 256 / BM), 256, 0, stream>>>(act16, wlT, b_lstm, nullptr, z, 2048, 4096);
    // 2) gates -> h (f16)
    lstm_gates_kernel<<<256, 256, 0, stream>>>(z, c_prev, h16);
    // 3) output_from_hidden, interface vector
    gemm_wmma_kernel<false><<<dim3(512 / BN, 2), 256, 0, stream>>>(h16, wohT, bo_h, nullptr, outh, 1024, 512);
    gemm_wmma_kernel<false><<<dim3(960 / BN, 2), 256, 0, stream>>>(h16, wintT, bint960, nullptr, pbuf, 1024, 960);
    // 4) addressing (pass 1 over memory)
    addressing_kernel<<<256, 1024, 0, stream>>>(pbuf, read_w_prev, write_w_prev, memory_prev, wread, wwrite);
    // 5) fused memory update + read_vec (pass 2 over memory)
    mem_update_kernel<<<256, 1024, 0, stream>>>(memory_prev, pbuf, wread, wwrite, out_mem, rv16);
    // 6) final output = rv @ Wout + b_out + output_from_hidden
    gemm_wmma_kernel<true><<<dim3(512 / BN, 2), 256, 0, stream>>>(rv16, woutT, b_out, outh, out_final, 512, 512);
}
